// MaskNet_73203422593254
// MI455X (gfx1250) — compile-verified, matrix operand order checked by reading, not run on hardware
//
#include <hip/hip_runtime.h>
#include <hip/hip_bf16.h>

#define NM    16
#define DMDIM 256
#define DIDIM 512
#define DSDIM 16
#define DCONVK 4
#define DTRDIM 16
#define ENCD  512
#define NSPKD 2
#define BB    2
#define LL    512
#define BT    (BB*LL)            // 1024 token rows
#define XD    (DTRDIM + 2*DSDIM) // 48

typedef __attribute__((ext_vector_type(16))) __bf16 v16bf;
typedef __attribute__((ext_vector_type(8)))  float  v8f;

union BF16Frag { uint4 u[2]; v16bf v; };

static __device__ __forceinline__ unsigned short f2bf(float f) {
    unsigned int u = __float_as_uint(f);
    u += 0x7FFFu + ((u >> 16) & 1u);            // round-to-nearest-even
    return (unsigned short)(u >> 16);
}

static __device__ __forceinline__ float block_sum_256(float v, float* sh) {
    __syncthreads();
    #pragma unroll
    for (int o = 16; o >= 1; o >>= 1) v += __shfl_xor(v, o, 32);
    int lane = threadIdx.x & 31, wv = threadIdx.x >> 5;
    if (lane == 0) sh[wv] = v;
    __syncthreads();
    float r = (threadIdx.x < 8) ? sh[threadIdx.x] : 0.f;
    #pragma unroll
    for (int o = 4; o >= 1; o >>= 1) r += __shfl_xor(r, o, 32);
    if (threadIdx.x == 0) sh[0] = r;
    __syncthreads();
    return sh[0];
}

// ---------------- f32 -> bf16 weight conversion ----------------
__global__ void cvt_bf16_kernel(const float* __restrict__ src,
                                unsigned short* __restrict__ dst, int n) {
    int i = blockIdx.x * blockDim.x + threadIdx.x;
    if (i < n) dst[i] = f2bf(src[i]);
}

// ---------------- channelwise LayerNorm over ENC, emits bf16 rows ----------------
__global__ void ln_kernel(const float* __restrict__ mix, const float* __restrict__ g,
                          const float* __restrict__ be, unsigned short* __restrict__ yb) {
    __shared__ float sh[8];
    int bt = blockIdx.x;               // 0..BT-1
    int b = bt >> 9, t = bt & (LL - 1);
    const float* base = mix + (size_t)b * ENCD * LL + t;   // x[b,t,e] = mix[b,e,t]
    int e0 = threadIdx.x, e1 = threadIdx.x + 256;
    float x0 = base[(size_t)e0 * LL];
    float x1 = base[(size_t)e1 * LL];
    float mu = block_sum_256(x0 + x1, sh) * (1.f / ENCD);
    float d0 = x0 - mu, d1 = x1 - mu;
    float var = block_sum_256(d0 * d0 + d1 * d1, sh) * (1.f / ENCD);
    float inv = rsqrtf(var + 1e-8f);
    unsigned short* out = yb + (size_t)bt * ENCD;
    out[e0] = f2bf(d0 * inv * g[e0] + be[e0]);
    out[e1] = f2bf(d1 * inv * g[e1] + be[e1]);
}

// ---------------- residual update + RMSNorm, emits bf16 rows ----------------
__global__ void resid_rms_kernel(const float* __restrict__ hidden, float* __restrict__ residual,
                                 const float* __restrict__ nw, unsigned short* __restrict__ hbf,
                                 int first) {
    __shared__ float sh[8];
    int bt = blockIdx.x, d = threadIdx.x;     // blockDim == DMDIM == 256
    size_t i = (size_t)bt * DMDIM + d;
    float r = hidden[i] + (first ? 0.f : residual[i]);
    residual[i] = r;
    float ss = block_sum_256(r * r, sh);
    float rms = rsqrtf(ss * (1.f / DMDIM) + 1e-5f);
    hbf[i] = f2bf(r * rms * nw[d]);
}

// ---------------- bf16 WMMA GEMM: C[M,N] = A[M,K] * W[N,K]^T (+C) ----------------
// Each wave computes a 16x(16*NT) strip: one A fragment feeds NT independent WMMA
// accumulator chains (A reuse + latency overlap). K loop is double-buffered (K%64==0);
// sched_barrier(0) pins the "load k+32 fragments" block above the "WMMA on k" block so
// the scheduler cannot re-serialize them into shared registers -> staggered loadcnt
// waits (loads in flight while the matrix pipe runs). K and the inter-row stride are
// compile-time, so all NT B-row addresses fold into instruction immediates of a single
// base pointer (two 64-bit address pairs total -> no spills). Fragments are two
// global_load_b128 per lane, matching the CDNA5 16-bit striping
// (lanes 0-15: K 0-7,16-23; lanes 16-31: K 8-15,24-31).
template<int NT, int K>
__global__ void gemm_bf16_wmma(const unsigned short* __restrict__ A,
                               const unsigned short* __restrict__ W,
                               float* __restrict__ C,
                               int M, int N, int accumulate) {
    int wave = threadIdx.x >> 5;
    int lane = threadIdx.x & 31;
    int nstrips = (N >> 4) / NT;
    int strip = blockIdx.x * (blockDim.x >> 5) + wave;
    if (strip >= (M >> 4) * nstrips) return;   // wave-uniform: EXEC stays all-ones
    int tm = strip / nstrips, sn = strip % nstrips;
    int mr = lane & 15, half = lane >> 4;

    const unsigned short* arow  = A + (size_t)(tm * 16 + mr) * K + half * 8;
    const unsigned short* brow0 = W + (size_t)(sn * NT * 16 + mr) * K + half * 8;

    int mbase = tm * 16 + half * 8;

    v8f acc[NT];
    #pragma unroll
    for (int j = 0; j < NT; ++j) {
        int n = (sn * NT + j) * 16 + mr;
        if (accumulate) {
            #pragma unroll
            for (int r = 0; r < 8; ++r) acc[j][r] = C[(size_t)(mbase + r) * N + n];
        } else {
            #pragma unroll
            for (int r = 0; r < 8; ++r) acc[j][r] = 0.f;
        }
    }

    auto loadA = [&](BF16Frag& f, int k0) {
        f.u[0] = *(const uint4*)(arow + k0);
        f.u[1] = *(const uint4*)(arow + k0 + 16);
    };
    auto loadB = [&](BF16Frag* f, int k0) {
        #pragma unroll
        for (int j = 0; j < NT; ++j) {      // j*16*K is a compile-time immediate
            f[j].u[0] = *(const uint4*)(brow0 + j * 16 * K + k0);
            f[j].u[1] = *(const uint4*)(brow0 + j * 16 * K + k0 + 16);
        }
    };

    BF16Frag a0, a1;
    BF16Frag b0[NT], b1[NT];
    loadA(a0, 0);
    loadB(b0, 0);
    for (int k0 = 0; k0 < K; k0 += 64) {       // K % 64 == 0 for all call sites
        loadA(a1, k0 + 32);
        loadB(b1, k0 + 32);
        if (k0 + 128 < K) {
            __builtin_prefetch(arow + k0 + 128, 0, 1);
            __builtin_prefetch(brow0 + k0 + 128, 0, 1);
        }
        __builtin_amdgcn_sched_barrier(0);     // keep next-step loads above these WMMAs
        #pragma unroll
        for (int j = 0; j < NT; ++j)
            acc[j] = __builtin_amdgcn_wmma_f32_16x16x32_bf16(
                false, a0.v, false, b0[j].v, (short)0, acc[j], false, false);
        if (k0 + 64 < K) {
            loadA(a0, k0 + 64);
            loadB(b0, k0 + 64);
        }
        __builtin_amdgcn_sched_barrier(0);     // same for the second half-step
        #pragma unroll
        for (int j = 0; j < NT; ++j)
            acc[j] = __builtin_amdgcn_wmma_f32_16x16x32_bf16(
                false, a1.v, false, b1[j].v, (short)0, acc[j], false, false);
    }

    #pragma unroll
    for (int j = 0; j < NT; ++j) {
        int n = (sn * NT + j) * 16 + mr;
        #pragma unroll
        for (int r = 0; r < 8; ++r) C[(size_t)(mbase + r) * N + n] = acc[j][r];
    }
}

// ---------------- depthwise causal/anticausal conv(4) + bias + SiLU ----------------
__global__ void conv_silu_kernel(const float* __restrict__ xz, const float* __restrict__ cw,
                                 const float* __restrict__ cb, float* __restrict__ xca,
                                 unsigned short* __restrict__ xcb, int back) {
    int idx = blockIdx.x * blockDim.x + threadIdx.x;       // over BT*DI
    if (idx >= BT * DIDIM) return;
    int d  = idx & (DIDIM - 1);
    int bt = idx >> 9;
    int t  = bt & (LL - 1), b = bt >> 9;
    float acc = cb[d];
    #pragma unroll
    for (int j = 0; j < DCONVK; ++j) {
        int tt = back ? (t + (DCONVK - 1) - j) : (t - (DCONVK - 1) + j);
        if (tt >= 0 && tt < LL)
            acc += xz[(size_t)(b * LL + tt) * (2 * DIDIM) + d] * cw[d * DCONVK + j];
    }
    float s = acc / (1.f + expf(-acc));
    xca[idx] = s;
    xcb[idx] = f2bf(s);
}

// ---------------- dt = softplus(xdbl[:, :16] @ dt_w^T + dt_b) ----------------
__global__ void dt_softplus_kernel(const float* __restrict__ xdbl, const float* __restrict__ dtw,
                                   const float* __restrict__ dtb, float* __restrict__ dt) {
    int idx = blockIdx.x * blockDim.x + threadIdx.x;       // over BT*DI
    if (idx >= BT * DIDIM) return;
    int d  = idx & (DIDIM - 1);
    int bt = idx >> 9;
    const float* xr = xdbl + (size_t)bt * XD;
    const float* wr = dtw + (size_t)d * DTRDIM;
    float a = dtb[d];
    #pragma unroll
    for (int r = 0; r < DTRDIM; ++r) a += xr[r] * wr[r];
    dt[idx] = (a > 20.f) ? a : log1pf(expf(a));
}

// ---------------- selective scan + D-skip + SiLU(z) gate, emits bf16 ----------------
// 16 lanes (s = state index) per (b, d) channel; __shfl_xor butterfly reduces over DS.
__global__ void scan_kernel(const float* __restrict__ dt, const float* __restrict__ xdbl,
                            const float* __restrict__ xca, const float* __restrict__ xz,
                            const float* __restrict__ Alog, const float* __restrict__ Dp,
                            unsigned short* __restrict__ ybf, int back) {
    int gid = (blockIdx.x * blockDim.x + threadIdx.x) >> 4;  // (b,d) group
    int s = threadIdx.x & 15;
    if (gid >= BB * DIDIM) return;
    int d = gid & (DIDIM - 1), b = gid >> 9;
    float a  = -expf(Alog[(size_t)d * DSDIM + s]);
    float Dd = Dp[d];
    float h = 0.f;
    for (int step = 0; step < LL; ++step) {
        int t  = back ? (LL - 1 - step) : step;
        int bt = b * LL + t;
        float dtv = dt[(size_t)bt * DIDIM + d];
        float xv  = xca[(size_t)bt * DIDIM + d];
        float Bm  = xdbl[(size_t)bt * XD + DTRDIM + s];
        float Cm  = xdbl[(size_t)bt * XD + DTRDIM + DSDIM + s];
        h = expf(dtv * a) * h + (dtv * xv) * Bm;
        float p = h * Cm;
        p += __shfl_xor(p, 1, 32);
        p += __shfl_xor(p, 2, 32);
        p += __shfl_xor(p, 4, 32);
        p += __shfl_xor(p, 8, 32);
        if (s == 0) {
            float y = p + Dd * xv;
            float z = xz[(size_t)bt * (2 * DIDIM) + DIDIM + d];
            y *= z / (1.f + expf(-z));
            ybf[(size_t)bt * DIDIM + d] = f2bf(y);
        }
    }
}

// ---------------- relu + [BT, 2*ENC] -> [NSPK, B, ENC, K] ----------------
__global__ void relu_transpose_kernel(const float* __restrict__ score, float* __restrict__ out) {
    int idx = blockIdx.x * blockDim.x + threadIdx.x;
    if (idx >= NSPKD * BB * ENCD * LL) return;
    int t   = idx & (LL - 1);
    int e   = (idx >> 9) & (ENCD - 1);
    int b   = (idx >> 18) & (BB - 1);
    int spk = idx >> 19;
    float v = score[(size_t)(b * LL + t) * (NSPKD * ENCD) + spk * ENCD + e];
    out[idx] = v > 0.f ? v : 0.f;
}

extern "C" void kernel_launch(void* const* d_in, const int* in_sizes, int n_in,
                              void* d_out, int out_size, void* d_ws, size_t ws_size,
                              hipStream_t stream) {
    const float* mix     = (const float*)d_in[0];
    const float* ln_g    = (const float*)d_in[1];
    const float* ln_b    = (const float*)d_in[2];
    const float* bneck_w = (const float*)d_in[3];
    const float* mask_w  = (const float*)d_in[4];
    const float* norm_w  = (const float*)d_in[5];
    const float* normf_w = (const float*)d_in[6];
    const float* in_proj[2]  = { (const float*)d_in[7],  (const float*)d_in[16] };
    const float* conv_w[2]   = { (const float*)d_in[8],  (const float*)d_in[17] };
    const float* conv_b[2]   = { (const float*)d_in[9],  (const float*)d_in[18] };
    const float* x_proj[2]   = { (const float*)d_in[10], (const float*)d_in[19] };
    const float* dt_w[2]     = { (const float*)d_in[11], (const float*)d_in[20] };
    const float* dt_b[2]     = { (const float*)d_in[12], (const float*)d_in[21] };
    const float* A_log[2]    = { (const float*)d_in[13], (const float*)d_in[22] };
    const float* Dparm[2]    = { (const float*)d_in[14], (const float*)d_in[23] };
    const float* out_proj[2] = { (const float*)d_in[15], (const float*)d_in[24] };

    // workspace carve-out (256B aligned)
    char* wsp = (char*)d_ws;
    size_t off = 0;
    auto alloc = [&](size_t bytes) -> void* {
        void* p = wsp + off;
        off = (off + bytes + 255) & ~(size_t)255;
        return p;
    };
    unsigned short* w_in[2], *w_xp[2], *w_op[2];
    for (int dir = 0; dir < 2; ++dir) {
        w_in[dir] = (unsigned short*)alloc((size_t)NM * 2 * DIDIM * DMDIM * 2);
        w_xp[dir] = (unsigned short*)alloc((size_t)NM * XD * DIDIM * 2);
        w_op[dir] = (unsigned short*)alloc((size_t)NM * DMDIM * DIDIM * 2);
    }
    unsigned short* w_bneck = (unsigned short*)alloc((size_t)DMDIM * ENCD * 2);
    unsigned short* w_mask  = (unsigned short*)alloc((size_t)NSPKD * ENCD * DMDIM * 2);
    unsigned short* ylnbf   = (unsigned short*)alloc((size_t)BT * ENCD * 2);
    float*          hidden  = (float*)alloc((size_t)BT * DMDIM * 4);
    float*          resid   = (float*)alloc((size_t)BT * DMDIM * 4);
    unsigned short* hbf     = (unsigned short*)alloc((size_t)BT * DMDIM * 2);
    float*          xz      = (float*)alloc((size_t)BT * 2 * DIDIM * 4);
    float*          xca     = (float*)alloc((size_t)BT * DIDIM * 4);
    unsigned short* xcb     = (unsigned short*)alloc((size_t)BT * DIDIM * 2);
    float*          xdbl    = (float*)alloc((size_t)BT * XD * 4);
    float*          dtbuf   = (float*)alloc((size_t)BT * DIDIM * 4);
    unsigned short* ybf     = (unsigned short*)alloc((size_t)BT * DIDIM * 2);
    float*          score   = (float*)alloc((size_t)BT * NSPKD * ENCD * 4);
    (void)ws_size;

    auto cvt = [&](const float* src, unsigned short* dst, int n) {
        cvt_bf16_kernel<<<(n + 255) / 256, 256, 0, stream>>>(src, dst, n);
    };
    // Strip GEMM dispatch over compile-time (NT, K):
    //   NT=4 when N%64==0 (N in {256,1024}); NT=3 covers N=48 exactly. K in {256,512}.
    auto gemm = [&](const unsigned short* A, const unsigned short* W, float* C,
                    int M, int N, int K, int acc) {
        if ((N & 63) == 0) {
            int strips = (M / 16) * (N / 64);
            int blocks = (strips + 7) / 8;
            if (K == 256)
                gemm_bf16_wmma<4, 256><<<blocks, 256, 0, stream>>>(A, W, C, M, N, acc);
            else
                gemm_bf16_wmma<4, 512><<<blocks, 256, 0, stream>>>(A, W, C, M, N, acc);
        } else {                                   // N == 48, K == 512
            int strips = (M / 16) * (N / 48);
            gemm_bf16_wmma<3, 512><<<(strips + 7) / 8, 256, 0, stream>>>(A, W, C, M, N, acc);
        }
    };

    // 1) convert all GEMM weights to bf16 once
    for (int dir = 0; dir < 2; ++dir) {
        cvt(in_proj[dir],  w_in[dir], NM * 2 * DIDIM * DMDIM);
        cvt(x_proj[dir],   w_xp[dir], NM * XD * DIDIM);
        cvt(out_proj[dir], w_op[dir], NM * DMDIM * DIDIM);
    }
    cvt(bneck_w, w_bneck, DMDIM * ENCD);
    cvt(mask_w,  w_mask,  NSPKD * ENCD * DMDIM);

    // 2) channelwise LN + bottleneck 1x1 conv
    ln_kernel<<<BT, 256, 0, stream>>>(mix, ln_g, ln_b, ylnbf);
    gemm(ylnbf, w_bneck, hidden, BT, DMDIM, ENCD, 0);

    // 3) 16 bidirectional Mamba layers
    const int NCH = BT * DIDIM;
    for (int i = 0; i < NM; ++i) {
        resid_rms_kernel<<<BT, 256, 0, stream>>>(hidden, resid,
                                                 norm_w + (size_t)i * DMDIM, hbf, i == 0);
        for (int dir = 0; dir < 2; ++dir) {
            gemm(hbf, w_in[dir] + (size_t)i * 2 * DIDIM * DMDIM, xz, BT, 2 * DIDIM, DMDIM, 0);
            conv_silu_kernel<<<(NCH + 255) / 256, 256, 0, stream>>>(
                xz, conv_w[dir] + (size_t)i * DIDIM * DCONVK,
                conv_b[dir] + (size_t)i * DIDIM, xca, xcb, dir);
            gemm(xcb, w_xp[dir] + (size_t)i * XD * DIDIM, xdbl, BT, XD, DIDIM, 0);
            dt_softplus_kernel<<<(NCH + 255) / 256, 256, 0, stream>>>(
                xdbl, dt_w[dir] + (size_t)i * DIDIM * DTRDIM,
                dt_b[dir] + (size_t)i * DIDIM, dtbuf);
            scan_kernel<<<(BB * DIDIM * 16 + 255) / 256, 256, 0, stream>>>(
                dtbuf, xdbl, xca, xz,
                A_log[dir] + (size_t)i * DIDIM * DSDIM,
                Dparm[dir] + (size_t)i * DIDIM, ybf, dir);
            gemm(ybf, w_op[dir] + (size_t)i * DMDIM * DIDIM, hidden, BT, DMDIM, DIDIM, dir);
        }
    }

    // 4) final residual + RMSNorm, mask head, relu + transpose
    resid_rms_kernel<<<BT, 256, 0, stream>>>(hidden, resid, normf_w, hbf, 0);
    gemm(hbf, w_mask, score, BT, NSPKD * ENCD, DMDIM, 0);
    relu_transpose_kernel<<<(NSPKD * BB * ENCD * LL + 255) / 256, 256, 0, stream>>>(score,
                                                                                    (float*)d_out);
}
